// Attention_88656714924481
// MI455X (gfx1250) — matrix operand order repaired and checked by reading, not compiled, not run on hardware
//
#include <hip/hip_runtime.h>
#include <hip/hip_bf16.h>

typedef __attribute__((ext_vector_type(16))) __bf16 bf16x16;
typedef __attribute__((ext_vector_type(8)))  float  f32x8;
typedef __attribute__((ext_vector_type(4)))  unsigned u32x4;
typedef __attribute__((ext_vector_type(8)))  unsigned u32x8;

union Frag16 { bf16x16 v; uint4 q[2]; };
union Pack8  { uint4 q; unsigned short s[8]; };

__device__ __forceinline__ __bf16 f2bf(float f) {
  union { float f; unsigned u; } x; x.f = f;
  unsigned r = x.u + 0x7FFFu + ((x.u >> 16) & 1u);   // round-to-nearest-even
  unsigned short h = (unsigned short)(r >> 16);
  __bf16 o; __builtin_memcpy(&o, &h, 2); return o;
}

// A-fragment (16x32 bf16, MxK) from row-major LDS slab. ISA 7.12.2:
// lanes 0-15 hold K=0..7 (v0-3) and K=16..23 (v4-7); lanes 16-31 hold +8.
__device__ __forceinline__ bf16x16 load_a_frag(const __bf16* base, int ld, int k0) {
  int lane = threadIdx.x & 31;
  int half = lane >> 4, m = lane & 15;
  const __bf16* p = base + m * ld + k0;
  Frag16 f;
  f.q[0] = *(const uint4*)(p + half * 8);
  f.q[1] = *(const uint4*)(p + 16 + half * 8);
  return f.v;
}

// B-fragment (32x16 bf16, KxN) from column-major LDS ([n][k], stride ld).
__device__ __forceinline__ bf16x16 load_b_frag(const __bf16* base, int ld) {
  int lane = threadIdx.x & 31;
  int half = lane >> 4, n = lane & 15;
  const uint4* p = (const uint4*)(base + n * ld + half * 8);
  Frag16 f;
  f.q[0] = p[0];
  f.q[1] = p[2];
  return f.v;
}

__device__ __forceinline__ f32x8 wmma_bf16(bf16x16 a, bf16x16 b, f32x8 c) {
  return __builtin_amdgcn_wmma_f32_16x16x32_bf16(false, a, false, b, (short)0, c, false, false);
}

// Generic LDS address -> 32-bit LDS byte offset (aperture maps addr[31:0]).
__device__ __forceinline__ unsigned lds_off(const void* p) {
  return (unsigned)(unsigned long long)p;
}

// ---------------------------------------------------------------------------
// Tensor Data Mover: build D# groups 0/1 (ISA 8.3/8.4) and issue
// tensor_load_to_lds with 2 SGPR groups (tensors up to 2D). data_size = 1B.
// tile_d1 == 0 -> 1D copy of tile_b0 bytes. Optional LDS row padding.
// Issue from ONE wave only; wait with s_wait_tensorcnt before the barrier.
// ---------------------------------------------------------------------------
__device__ __forceinline__ void tdm_load(unsigned lds_addr, const void* gsrc,
                                         unsigned tile_b0, unsigned tile_d1,
                                         unsigned long long stride0_b,
                                         unsigned tensor_b0, unsigned tensor_d1,
                                         unsigned pad_en, unsigned pad_int,
                                         unsigned pad_amt) {
  unsigned long long ga = (unsigned long long)gsrc;
  u32x4 g0 = {
    1u,                                         // count=1, user descriptor
    lds_addr,                                   // lds_addr (bytes)
    (unsigned)ga,                               // global_addr[31:0]
    (unsigned)(ga >> 32) | (2u << 30)           // global_addr[56:32] | type=2
  };
  u32x8 g1 = {
    (pad_en << 20) | (pad_int << 22) | (pad_amt << 25),      // mask=0, 1B elems
    (tensor_b0 & 0xFFFFu) << 16,                             // tensor_dim0 lo
    (tensor_b0 >> 16) | ((tensor_d1 & 0xFFFFu) << 16),       // dim0 hi | dim1 lo
    ((tensor_d1 >> 16) & 0xFFFFu) | ((tile_b0 & 0xFFFFu) << 16), // dim1 hi|tile0
    tile_d1 & 0xFFFFu,                                       // tile_dim1, dim2=0
    (unsigned)stride0_b,                                     // stride0[31:0]
    (unsigned)(stride0_b >> 32) & 0xFFFFu,                   // stride0[47:32]
    0u                                                       // stride1 hi
  };
  asm volatile("tensor_load_to_lds %0, %1" :: "s"(g0), "s"(g1) : "memory");
}

// ---------------------------------------------------------------------------
// bf16 GEMM: C[M,N] (fp32) = A[M,K] x B[K,N], row-major. 128x128x64 tiles,
// 8 waves (4x2); wave = 32x64 slab = 8 C tiles -> 8 WMMA per 6 frag loads.
// A tile via TDM (2D, padded rows); B tile manual transpose (pair-packed b32).
// ---------------------------------------------------------------------------
__global__ __launch_bounds__(256) void gemm_bf16_kernel(
    const __bf16* __restrict__ A, const __bf16* __restrict__ B,
    float* __restrict__ C, int M, int N, int K) {
  constexpr int LDA = 72, LDB = 72;  // 144B rows = 128B data + 16B TDM pad
  __shared__ __align__(16) __bf16 As[128 * LDA];
  __shared__ __align__(16) __bf16 Bs[128 * LDB];   // column-major [n][k]
  int t = threadIdx.x;
  int wave = t >> 5;
  int wr = wave >> 1;            // 0..3 : 32-row slab
  int wc = wave & 1;             // 0..1 : 64-col slab
  int rowBase = blockIdx.y * 128;
  int colBase = blockIdx.x * 128;
  f32x8 acc[2][4];
#pragma unroll
  for (int i = 0; i < 2; i++)
#pragma unroll
    for (int j = 0; j < 4; j++) acc[i][j] = (f32x8){};
  unsigned asLds = lds_off(As);

  for (int k0 = 0; k0 < K; k0 += 64) {
    if (t < 32)   // TDM: 128 rows x 128B, row stride K*2B, +16B/128B LDS pad
      tdm_load(asLds, A + (size_t)rowBase * K + k0,
               128u, 128u, (unsigned long long)K * 2,
               (unsigned)K * 2, (unsigned)M,
               1u, 4u /*128B interval*/, 3u /*16B pad*/);
    { // B tile 64x128 -> Bs[n][k], two k-rows packed per b32 store
      int r2 = t >> 3;                 // k rows 2*r2, 2*r2+1
      int nc = (t & 7) * 16;
      const __bf16* g0p = B + (size_t)(k0 + 2 * r2) * N + colBase + nc;
      const __bf16* g1p = g0p + N;
      Pack8 a0, a1, b0, b1;
      a0.q = *(const uint4*)(g0p);     a1.q = *(const uint4*)(g0p + 8);
      b0.q = *(const uint4*)(g1p);     b1.q = *(const uint4*)(g1p + 8);
#pragma unroll
      for (int i = 0; i < 8; i++) {
        *(unsigned*)(Bs + (nc + i) * LDB + 2 * r2) =
            (unsigned)a0.s[i] | ((unsigned)b0.s[i] << 16);
        *(unsigned*)(Bs + (nc + 8 + i) * LDB + 2 * r2) =
            (unsigned)a1.s[i] | ((unsigned)b1.s[i] << 16);
      }
    }
    if (k0 + 64 < K)
      __builtin_prefetch(B + (size_t)(k0 + 64 + (t >> 3)) * N + colBase, 0, 1);
    if (t < 32) __builtin_amdgcn_s_wait_tensorcnt(0);
    __syncthreads();
#pragma unroll
    for (int ks = 0; ks < 64; ks += 32) {
      bf16x16 a0f = load_a_frag(As + (wr * 32)      * LDA, LDA, ks);
      bf16x16 a1f = load_a_frag(As + (wr * 32 + 16) * LDA, LDA, ks);
      bf16x16 bf[4];
#pragma unroll
      for (int j = 0; j < 4; j++)
        bf[j] = load_b_frag(Bs + (wc * 64 + j * 16) * LDB + ks, LDB);
#pragma unroll
      for (int j = 0; j < 4; j++) {
        acc[0][j] = wmma_bf16(a0f, bf[j], acc[0][j]);
        acc[1][j] = wmma_bf16(a1f, bf[j], acc[1][j]);
      }
    }
    __syncthreads();
  }

  int lane = t & 31, half = lane >> 4, lr = lane & 15;
#pragma unroll
  for (int i = 0; i < 2; i++)
#pragma unroll
    for (int j = 0; j < 4; j++)
#pragma unroll
      for (int v = 0; v < 8; v++) {
        int row = rowBase + wr * 32 + i * 16 + half * 8 + v;
        int col = colBase + wc * 64 + j * 16 + lr;
        C[(size_t)row * N + col] = acc[i][j][v];
      }
}

// ---------------------------------------------------------------------------
__global__ void f32_to_bf16_kernel(const float* __restrict__ src,
                                   __bf16* __restrict__ dst, size_t n) {
  for (size_t i = blockIdx.x * (size_t)blockDim.x + threadIdx.x; i < n;
       i += (size_t)gridDim.x * blockDim.x)
    dst[i] = f2bf(src[i]);
}

// RoPE + transpose: src fp32 [B,S,nh,128] -> dst bf16 [B,nh,S,128]
__global__ void rope_transpose_kernel(const float* __restrict__ src,
                                      const float* __restrict__ fc,
                                      const float* __restrict__ fs,
                                      __bf16* __restrict__ dst, int nh) {
  const int S = 2048, HD = 128;
  size_t total = (size_t)2 * S * nh * 64;
  for (size_t idx = blockIdx.x * (size_t)blockDim.x + threadIdx.x; idx < total;
       idx += (size_t)gridDim.x * blockDim.x) {
    int i = (int)(idx & 63);
    size_t r = idx >> 6;
    int h = (int)(r % nh); r /= nh;
    int s = (int)(r % S);
    int b = (int)(r / S);
    const float* p = src + ((size_t)(b * S + s) * nh + h) * HD + 2 * i;
    float p1 = p[0], p2 = p[1];
    float c = fc[s * 64 + i], sn = fs[s * 64 + i];
    __bf16* q = dst + ((size_t)(b * nh + h) * S + s) * HD + 2 * i;
    q[0] = f2bf(p1 * c - p2 * sn);
    q[1] = f2bf(p1 * sn + p2 * c);
  }
}

__global__ void v_transpose_kernel(const float* __restrict__ src,
                                   __bf16* __restrict__ dst, int nh) {
  const int S = 2048, HD = 128;
  size_t total = (size_t)2 * S * nh * HD;
  for (size_t idx = blockIdx.x * (size_t)blockDim.x + threadIdx.x; idx < total;
       idx += (size_t)gridDim.x * blockDim.x) {
    int d = (int)(idx & (HD - 1));
    size_t r = idx >> 7;
    int h = (int)(r % nh); r /= nh;
    int s = (int)(r % S);
    int b = (int)(r / S);
    dst[((size_t)(b * nh + h) * S + s) * HD + d] =
        f2bf(src[((size_t)(b * S + s) * nh + h) * HD + d]);
  }
}

// ---------------------------------------------------------------------------
// Causal flash attention. Q [B,16,S,128], K/V [B,8,S,128] bf16 (GQA 2:1),
// out bf16 [B,S,2048]. 4 waves; wave w owns q rows w*16..+15 of a 64-row tile.
// Q and K tiles are contiguous -> 1D TDM loads; V transposed manually.
// ---------------------------------------------------------------------------
__global__ __launch_bounds__(128) void flash_attn_kernel(
    const __bf16* __restrict__ Q, const __bf16* __restrict__ Kt,
    const __bf16* __restrict__ V, __bf16* __restrict__ Out) {
  constexpr int D = 128, S = 2048, H = 16, KVH = 8;
  constexpr int LDQ = 128, LDK = 128, LDV = 40, LDP = 40;
  __shared__ __align__(16) __bf16 Qs[64 * LDQ];
  __shared__ __align__(16) __bf16 Ks[32 * LDK];     // row-major [key][d]
  __shared__ __align__(16) __bf16 Vt[128 * LDV];    // transposed [d][key]
  __shared__ __align__(16) __bf16 Ps[4][16 * LDP];  // per-wave P tile

  int t = threadIdx.x;
  int wave = t >> 5;
  int lane = t & 31, half = lane >> 4, lr = lane & 15;
  int b = blockIdx.x / H, h = blockIdx.x % H;
  int kvh = h >> 1;
  int q0 = blockIdx.y * 64;
  int qrow0 = wave * 16;

  const __bf16* Qbase = Q  + ((size_t)(b * H + h) * S + q0) * D;
  const __bf16* Kbase = Kt + ((size_t)(b * KVH + kvh) * S) * D;
  const __bf16* Vbase = V  + ((size_t)(b * KVH + kvh) * S) * D;

  if (t < 32)   // Q tile: 64x128 bf16 = 16KB contiguous -> 1D TDM
    tdm_load(lds_off(Qs), Qbase, 16384u, 0u, 16384ull, 16384u, 1u, 0u, 0u, 0u);

  f32x8 O[8];
#pragma unroll
  for (int i = 0; i < 8; i++) O[i] = (f32x8){};
  float mrow[8], lrow[8];
#pragma unroll
  for (int v = 0; v < 8; v++) { mrow[v] = -INFINITY; lrow[v] = 0.f; }
  const float scale = 0.08838834764831845f;   // 1/sqrt(128)

  int nkt = blockIdx.y * 2 + 2;               // causal: keys up to q-tile end
  for (int kt = 0; kt < nkt; kt++) {
    int kb = kt * 32;
    __syncthreads();
    if (t < 32)  // K tile: 32x128 bf16 = 8KB contiguous -> 1D TDM
      tdm_load(lds_off(Ks), Kbase + (size_t)kb * D, 8192u, 0u, 8192ull,
               8192u, 1u, 0u, 0u, 0u);
    { // V tile 32x128 -> Vt[d][key], two keys packed per b32 store
      int r2 = t >> 3;                 // key rows 2*r2, 2*r2+1
      int nc = (t & 7) * 16;
      const __bf16* g0p = Vbase + (size_t)(kb + 2 * r2) * D + nc;
      const __bf16* g1p = g0p + D;
      Pack8 a0, a1, b0, b1;
      a0.q = *(const uint4*)(g0p);     a1.q = *(const uint4*)(g0p + 8);
      b0.q = *(const uint4*)(g1p);     b1.q = *(const uint4*)(g1p + 8);
#pragma unroll
      for (int i = 0; i < 8; i++) {
        *(unsigned*)(Vt + (nc + i) * LDV + 2 * r2) =
            (unsigned)a0.s[i] | ((unsigned)b0.s[i] << 16);
        *(unsigned*)(Vt + (nc + 8 + i) * LDV + 2 * r2) =
            (unsigned)a1.s[i] | ((unsigned)b1.s[i] << 16);
      }
    }
    if (t < 32) __builtin_amdgcn_s_wait_tensorcnt(0);
    __syncthreads();

    // S = Q * K^T  (16 q-rows x 32 keys)
    f32x8 s0 = {}, s1 = {};
#pragma unroll
    for (int ks = 0; ks < 4; ks++) {
      bf16x16 a  = load_a_frag(Qs + qrow0 * LDQ, LDQ, ks * 32);
      bf16x16 b0 = load_b_frag(Ks + ks * 32, LDK);            // keys 0..15
      bf16x16 b1 = load_b_frag(Ks + 16 * LDK + ks * 32, LDK); // keys 16..31
      s0 = wmma_bf16(a, b0, s0);
      s1 = wmma_bf16(a, b1, s1);
    }

    // Online softmax (C layout: row v+half*8 lives in one 16-lane group)
#pragma unroll
    for (int v = 0; v < 8; v++) {
      int q = q0 + qrow0 + half * 8 + v;
      float e0 = s0[v] * scale, e1 = s1[v] * scale;
      if (kb + lr > q)      e0 = -INFINITY;
      if (kb + 16 + lr > q) e1 = -INFINITY;
      float tm = fmaxf(e0, e1);
      tm = fmaxf(tm, __shfl_xor(tm, 1, 16));
      tm = fmaxf(tm, __shfl_xor(tm, 2, 16));
      tm = fmaxf(tm, __shfl_xor(tm, 4, 16));
      tm = fmaxf(tm, __shfl_xor(tm, 8, 16));
      float mnew = fmaxf(mrow[v], tm);
      float corr = __expf(mrow[v] - mnew);
      float p0 = __expf(e0 - mnew);
      float p1 = __expf(e1 - mnew);
      float rs = p0 + p1;
      rs += __shfl_xor(rs, 1, 16);
      rs += __shfl_xor(rs, 2, 16);
      rs += __shfl_xor(rs, 4, 16);
      rs += __shfl_xor(rs, 8, 16);
      lrow[v] = lrow[v] * corr + rs;
      mrow[v] = mnew;
#pragma unroll
      for (int nt = 0; nt < 8; nt++) O[nt][v] *= corr;
      Ps[wave][(half * 8 + v) * LDP + lr]      = f2bf(p0);
      Ps[wave][(half * 8 + v) * LDP + 16 + lr] = f2bf(p1);
    }
    asm volatile("s_wait_dscnt 0" ::: "memory");  // wave-local LDS transpose

    // O += P * V  (16 x 32) x (32 x 128)
    bf16x16 pa = load_a_frag(&Ps[wave][0], LDP, 0);
#pragma unroll
    for (int nt = 0; nt < 8; nt++) {
      bf16x16 bv = load_b_frag(Vt + nt * 16 * LDV, LDV);
      O[nt] = wmma_bf16(pa, bv, O[nt]);
    }
  }

  // epilogue: O/l -> bf16, layout [B,S,H*D]
  __bf16* outBase =
      Out + ((size_t)b * S + q0 + qrow0 + half * 8) * (H * D) + h * D + lr;
#pragma unroll
  for (int v = 0; v < 8; v++) {
    float inv = 1.0f / lrow[v];
#pragma unroll
    for (int nt = 0; nt < 8; nt++)
      outBase[(size_t)v * (H * D) + nt * 16] = f2bf(O[nt][v] * inv);
  }
}

// ---------------------------------------------------------------------------
extern "C" void kernel_launch(void* const* d_in, const int* in_sizes, int n_in,
                              void* d_out, int out_size, void* d_ws,
                              size_t ws_size, hipStream_t stream) {
  const float* x  = (const float*)d_in[0];
  const float* fc = (const float*)d_in[1];
  const float* fs = (const float*)d_in[2];
  const float* wq = (const float*)d_in[3];
  const float* wk = (const float*)d_in[4];
  const float* wv = (const float*)d_in[5];
  const float* wo = (const float*)d_in[6];
  float* out = (float*)d_out;

  const size_t B = 2, S = 2048, DIM = 2048, H = 16, KVH = 8, HD = 128;
  const size_t nX  = B * S * DIM;
  const size_t nWq = DIM * DIM;
  const size_t nWk = DIM * KVH * HD;
  const size_t nKV = B * S * KVH * HD;

  char* ws = (char*)d_ws;
  size_t off = 0;
  auto carve = [&](size_t bytes) -> void* {
    void* p = ws + off;
    off += (bytes + 255) & ~(size_t)255;
    return p;
  };
  __bf16* x_bf  = (__bf16*)carve(nX * 2);
  __bf16* wq_bf = (__bf16*)carve(nWq * 2);
  __bf16* wk_bf = (__bf16*)carve(nWk * 2);
  __bf16* wv_bf = (__bf16*)carve(nWk * 2);
  __bf16* wo_bf = (__bf16*)carve(nWq * 2);
  float*  qf    = (float*)carve(nX * 4);
  float*  kf    = (float*)carve(nKV * 4);
  float*  vf    = (float*)carve(nKV * 4);
  __bf16* Qbf   = (__bf16*)carve(nX * 2);
  __bf16* Kbf   = (__bf16*)carve(nKV * 2);
  __bf16* Vbf   = (__bf16*)carve(nKV * 2);
  __bf16* Abf   = (__bf16*)qf;   // reuse dead Q-fp32 region for attn output

  f32_to_bf16_kernel<<<2048, 256, 0, stream>>>(x,  x_bf,  nX);
  f32_to_bf16_kernel<<<1024, 256, 0, stream>>>(wq, wq_bf, nWq);
  f32_to_bf16_kernel<<<512,  256, 0, stream>>>(wk, wk_bf, nWk);
  f32_to_bf16_kernel<<<512,  256, 0, stream>>>(wv, wv_bf, nWk);
  f32_to_bf16_kernel<<<1024, 256, 0, stream>>>(wo, wo_bf, nWq);

  dim3 gQ(DIM / 128, (B * S) / 128);            // (16, 32)
  dim3 gK((KVH * HD) / 128, (B * S) / 128);     // (8, 32)
  gemm_bf16_kernel<<<gQ, 256, 0, stream>>>(x_bf, wq_bf, qf,
                                           (int)(B * S), (int)DIM, (int)DIM);
  gemm_bf16_kernel<<<gK, 256, 0, stream>>>(x_bf, wk_bf, kf,
                                           (int)(B * S), (int)(KVH * HD), (int)DIM);
  gemm_bf16_kernel<<<gK, 256, 0, stream>>>(x_bf, wv_bf, vf,
                                           (int)(B * S), (int)(KVH * HD), (int)DIM);

  rope_transpose_kernel<<<2048, 256, 0, stream>>>(qf, fc, fs, Qbf, (int)H);
  rope_transpose_kernel<<<1024, 256, 0, stream>>>(kf, fc, fs, Kbf, (int)KVH);
  v_transpose_kernel<<<2048, 256, 0, stream>>>(vf, Vbf, (int)KVH);

  flash_attn_kernel<<<dim3(B * H, S / 64), 128, 0, stream>>>(Qbf, Kbf, Vbf, Abf);

  gemm_bf16_kernel<<<gQ, 256, 0, stream>>>(Abf, wo_bf, out,
                                           (int)(B * S), (int)DIM, (int)DIM);
}
